// WindowAttention_21260088115593
// MI455X (gfx1250) — matrix operand order repaired and checked by reading, not compiled, not run on hardware
//
#include <hip/hip_runtime.h>
#include <stdint.h>

typedef __attribute__((ext_vector_type(16))) _Float16 v16h;
typedef __attribute__((ext_vector_type(8)))  float    v8f;
typedef uint32_t u32;

union Frag16 { v16h v; _Float16 h[16]; u32 u[8]; };
union U32H   { u32 u;  _Float16 h[2]; };

__device__ __forceinline__ int kbaseA(int j, int half) {
  // contiguous k order used here for A and B: VGPR j holds K = half*16 + j*2, +1
  return (half << 4) + (j << 1);
}

__device__ __forceinline__ v8f wmma_f16(v16h a, v16h b, v8f c) {
  return __builtin_amdgcn_wmma_f32_16x16x32_f16(false, a, false, b, (short)0, c, false, false);
}

// LDS byte offset of a generic shared pointer: per CDNA5 aperture rules the
// low 32 bits of a generic LDS address are the LDS offset (addr[31:0]).
__device__ __forceinline__ u32 lds_off(const void* p) {
  return (u32)(uintptr_t)p;
}

// Async global -> LDS copy of 16 bytes (per-lane addresses, ASYNCcnt tracked).
__device__ __forceinline__ void async_copy_b128(u32 lds_byte_off, const void* gptr) {
  asm volatile("global_load_async_to_lds_b128 %0, %1, off"
               :: "v"(lds_byte_off), "v"((uint64_t)(uintptr_t)gptr)
               : "memory");
}
__device__ __forceinline__ void wait_asynccnt0() {
  asm volatile("s_wait_asynccnt 0x0" ::: "memory");
}

// ---------------------------------------------------------------------------
// Kernel 1: QKV GEMM.  X (Mtot x 512, f32, optional row remap for pooled path)
//           x W_qkv (512 x 1536, f32) + b -> q/k/v f16 buffers [row][512].
// block 256 = 8 waves, each wave owns 16 rows; 64 output cols per block.
// ---------------------------------------------------------------------------
__global__ __launch_bounds__(256) void qkv_gemm_kernel(
    const float* __restrict__ X, const float* __restrict__ W,
    const float* __restrict__ bias,
    _Float16* __restrict__ qo, _Float16* __restrict__ ko, _Float16* __restrict__ vo,
    int Mtot, int n_base, int remap)
{
  __shared__ _Float16 s_w[64 * 32];   // [col][k] transposed -> u32 B-frag reads
  const int tid = threadIdx.x, lane = tid & 31, wave = tid >> 5;
  const int half = lane >> 4, l15 = lane & 15;
  const int m0 = blockIdx.x * 128 + wave * 16;
  const int n0 = n_base + blockIdx.y * 64;

  const int mrow = m0 + l15;
  const bool mval = mrow < Mtot;
  long srow = 0;
  if (mval) {
    if (remap) { int t = mrow / 144, rem = mrow % 144; srow = (long)rem * 5 + t; }
    else       { srow = mrow; }
  }
  const float* xr = X + (size_t)srow * 512;

  Frag16 a[16];
#pragma unroll
  for (int kc = 0; kc < 16; ++kc)
#pragma unroll
    for (int j = 0; j < 8; ++j) {
      int kb = kc * 32 + kbaseA(j, half);
      float f0 = mval ? xr[kb]     : 0.f;
      float f1 = mval ? xr[kb + 1] : 0.f;
      a[kc].h[2 * j]     = (_Float16)f0;
      a[kc].h[2 * j + 1] = (_Float16)f1;
    }

  v8f zero = {};
  v8f acc[4];
#pragma unroll
  for (int nt = 0; nt < 4; ++nt) acc[nt] = zero;

  for (int kc = 0; kc < 16; ++kc) {
    __syncthreads();
    // stage W chunk transposed: 16 threads cover one 64-float row (b128 loads)
#pragma unroll
    for (int i = tid; i < 512; i += 256) {
      int kk = i >> 4;            // 0..31
      int nn = (i & 15) * 4;      // 0..60
      float4 w4 = *(const float4*)&W[(size_t)(kc * 32 + kk) * 1536 + (n0 + nn)];
      s_w[(nn + 0) * 32 + kk] = (_Float16)w4.x;
      s_w[(nn + 1) * 32 + kk] = (_Float16)w4.y;
      s_w[(nn + 2) * 32 + kk] = (_Float16)w4.z;
      s_w[(nn + 3) * 32 + kk] = (_Float16)w4.w;
    }
    __syncthreads();
#pragma unroll
    for (int nt = 0; nt < 4; ++nt) {
      Frag16 b;
      const u32* wp = (const u32*)(s_w + (nt * 16 + l15) * 32);
#pragma unroll
      for (int j = 0; j < 8; ++j) b.u[j] = wp[half * 8 + j];
      acc[nt] = wmma_f16(a[kc].v, b.v, acc[nt]);
    }
  }

#pragma unroll
  for (int nt = 0; nt < 4; ++nt) {
    int n = n0 + nt * 16 + l15;
    float bn = bias[n];
#pragma unroll
    for (int r = 0; r < 8; ++r) {
      int m = m0 + r + 8 * half;
      if (m < Mtot) {
        _Float16 hv = (_Float16)(acc[nt][r] + bn);
        size_t row = (size_t)m * 512;
        if (n < 512)       { if (qo) qo[row + n] = hv; }
        else if (n < 1024) { ko[row + (n - 512)] = hv; }
        else               { vo[row + (n - 1024)] = hv; }
      }
    }
  }
}

// ---------------------------------------------------------------------------
// Kernel 2: flash-style window attention.  One block per (window, head).
// 512 threads = 16 waves; waves 0..14 own 16-row M tiles (225 queries -> 240).
// 33 chunks of 32 keys (1050 -> 1056, pad bias -1e30).
// K and V tiles are staged with global_load_async_to_lds_b128 (ASYNCcnt).
// ---------------------------------------------------------------------------
__global__ __launch_bounds__(512) void attn_kernel(
    const _Float16* __restrict__ qg, const _Float16* __restrict__ kgb,
    const _Float16* __restrict__ vgb, const _Float16* __restrict__ kp,
    const _Float16* __restrict__ vp, _Float16* __restrict__ ao)
{
  constexpr float SCALE = 0.08838834764831845f;   // 128^-0.5
  __shared__ _Float16 s_k[32 * 128];      // [key][ch]
  __shared__ _Float16 s_vstage[32 * 128]; // [key][ch] async-staged V
  __shared__ _Float16 s_vt[128 * 32];     // [ch][key]  (transposed V)
  __shared__ _Float16 s_p[15 * 16 * 32];
  __shared__ float    s_bias[32];
  __shared__ int      s_vir[120];

  const int tid = threadIdx.x, lane = tid & 31, wave = tid >> 5;
  const int half = lane >> 4, l15 = lane & 15;
  const int win = blockIdx.x, head = blockIdx.y;
  const int wy = win / 12, wx = win % 12;

  // Build VIR table (order matches np.nonzero of stacked tl/tr/bl/br masks).
  if (tid < 120) {
    int cnt = 0, rec = 0;
    for (int j = 0; j < 180; ++j) {
      int q = j / 45, p = j % 45, iy = p / 9, ix = p % 9;
      bool vld;
      if (q == 0)      vld = (iy >= 3) || (ix >= 5);
      else if (q == 1) vld = (iy >= 3) || (ix < 4);
      else if (q == 2) vld = (iy < 2)  || (ix >= 5);
      else             vld = (iy < 2)  || (ix < 4);
      if (vld) { if (cnt == tid) rec = (q << 8) | (iy << 4) | ix; ++cnt; }
    }
    s_vir[tid] = rec;
  }

  // Q fragments (4 chunks of K=32 covering HD=128)
  const int mq = wave * 16 + l15;
  Frag16 qf[4];
  if (mq < 225) {
    int t = mq / 45, p = mq % 45, iy = p / 9, ix = p % 9;
    size_t pix = (size_t)(t * 60 + wy * 5 + iy) * 108 + (wx * 9 + ix);
    const u32* qr = (const u32*)(qg + pix * 512 + head * 128);
#pragma unroll
    for (int kc = 0; kc < 4; ++kc)
#pragma unroll
      for (int j = 0; j < 8; ++j)
        qf[kc].u[j] = qr[kc * 16 + (kbaseA(j, half) >> 1)];
  } else {
#pragma unroll
    for (int kc = 0; kc < 4; ++kc)
#pragma unroll
      for (int j = 0; j < 8; ++j) qf[kc].u[j] = 0u;
  }

  v8f zero = {};
  v8f acc[8];
#pragma unroll
  for (int ct = 0; ct < 8; ++ct) acc[ct] = zero;
  float mrun[8], lrun[8];
#pragma unroll
  for (int r = 0; r < 8; ++r) { mrun[r] = -1e30f; lrun[r] = 0.f; }

  const int slot = tid >> 4, cg = tid & 15, c0 = cg * 8;

  for (int c32 = 0; c32 < 33; ++c32) {
    __syncthreads();
    // ---- cooperative gather of 32 keys ----
    {
      int kgi = c32 * 32 + slot;
      const _Float16* ks = nullptr; const _Float16* vs = nullptr;
      float bb = 0.f;
      if (kgi < 225) {                       // direct window keys
        int t = kgi / 45, p = kgi % 45, iy = p / 9, ix = p % 9;
        size_t po = ((size_t)(t * 60 + wy * 5 + iy) * 108 + (wx * 9 + ix)) * 512 + head * 128;
        ks = kgb + po; vs = vgb + po;
      } else if (kgi < 825) {                // rolled keys via VIR
        int r = kgi - 225, t = r / 120, vj = r % 120;
        int rec = s_vir[vj];
        int q = rec >> 8, iy = (rec >> 4) & 15, ix = rec & 15;
        int sy = (q < 2) ? -2 : 2;
        int sx = (q & 1) ? 4 : -4;
        int y = (wy * 5 + iy - sy + 60) % 60;
        int x = (wx * 9 + ix - sx + 108) % 108;
        size_t po = ((size_t)(t * 60 + y) * 108 + x) * 512 + head * 128;
        ks = kgb + po; vs = vgb + po;
      } else if (kgi < 1050) {               // pooled keys (zero-pad + -100)
        int r = kgi - 825, t = r / 45, p = r % 45, ky = p / 9, kx = p % 9;
        int py = wy + ky - 2, px = wx + kx - 4;
        if (py >= 0 && py < 12 && px >= 0 && px < 12) {
          size_t po = ((size_t)(t * 12 + py) * 12 + px) * 512 + head * 128;
          ks = kp + po; vs = vp + po;
        } else bb = -100.f;
      } else bb = -1e30f;                    // pad to 1056

      if (ks) {
        // async 16B copies: K row-major, V into staging for transpose
        async_copy_b128(lds_off(s_k + slot * 128 + c0), ks + c0);
        async_copy_b128(lds_off(s_vstage + slot * 128 + c0), vs + c0);
      } else {
        u32* kd = (u32*)(s_k + slot * 128);
#pragma unroll
        for (int u = 0; u < 4; ++u) {
          kd[(c0 >> 1) + u] = 0u;
          s_vt[(c0 + 2 * u) * 32 + slot]     = (_Float16)0.f;
          s_vt[(c0 + 2 * u + 1) * 32 + slot] = (_Float16)0.f;
        }
      }
      if (cg == 0) s_bias[slot] = bb;

      wait_asynccnt0();   // this wave's async copies have landed in LDS
      if (ks) {
        // transpose the 8 V halves this thread loaded itself
        const u32* vw = (const u32*)(s_vstage + slot * 128 + c0);
#pragma unroll
        for (int u = 0; u < 4; ++u) {
          U32H t2; t2.u = vw[u];
          s_vt[(c0 + 2 * u) * 32 + slot]     = t2.h[0];
          s_vt[(c0 + 2 * u + 1) * 32 + slot] = t2.h[1];
        }
      }
    }
    __syncthreads();

    if (wave < 15) {
      // ---- S = Q * K^T (two 16-key subtiles) ----
      v8f S0 = zero, S1 = zero;
#pragma unroll
      for (int kc = 0; kc < 4; ++kc) {
        Frag16 b0, b1;
        const u32* kt0 = (const u32*)(s_k + l15 * 128);
        const u32* kt1 = (const u32*)(s_k + (16 + l15) * 128);
#pragma unroll
        for (int j = 0; j < 8; ++j) {
          int ci = kc * 16 + half * 8 + j;
          b0.u[j] = kt0[ci];
          b1.u[j] = kt1[ci];
        }
        S0 = wmma_f16(qf[kc].v, b0.v, S0);
        S1 = wmma_f16(qf[kc].v, b1.v, S1);
      }
      float bs0 = s_bias[l15], bs1 = s_bias[16 + l15];
      float p0[8], p1[8];
#pragma unroll
      for (int r = 0; r < 8; ++r) {
        float sA = S0[r] * SCALE + bs0;
        float sB = S1[r] * SCALE + bs1;
        float cm = fmaxf(sA, sB);
#pragma unroll
        for (int d2 = 1; d2 < 16; d2 <<= 1) cm = fmaxf(cm, __shfl_xor(cm, d2, 32));
        float mn = fmaxf(mrun[r], cm);
        float alpha = __expf(mrun[r] - mn);
        float e0 = __expf(sA - mn);
        float e1 = __expf(sB - mn);
        float rs = e0 + e1;
#pragma unroll
        for (int d2 = 1; d2 < 16; d2 <<= 1) rs += __shfl_xor(rs, d2, 32);
        lrun[r] = lrun[r] * alpha + rs;
        mrun[r] = mn;
        p0[r] = e0; p1[r] = e1;
#pragma unroll
        for (int ct = 0; ct < 8; ++ct) acc[ct][r] *= alpha;
      }
      // ---- restage P (C-layout -> A-layout) through per-wave LDS ----
      _Float16* pr = s_p + wave * 512;
#pragma unroll
      for (int r = 0; r < 8; ++r) {
        int ml = r + 8 * half;
        pr[ml * 32 + l15]      = (_Float16)p0[r];
        pr[ml * 32 + 16 + l15] = (_Float16)p1[r];
      }
      asm volatile("" ::: "memory");   // keep LDS stores before re-reads
      Frag16 pf;
#pragma unroll
      for (int j = 0; j < 8; ++j)
        pf.u[j] = *(const u32*)(pr + l15 * 32 + kbaseA(j, half));
      // ---- acc += P * V ----
#pragma unroll
      for (int ct = 0; ct < 8; ++ct) {
        Frag16 bv;
        const u32* vt = (const u32*)(s_vt + (ct * 16 + l15) * 32);
#pragma unroll
        for (int j = 0; j < 8; ++j) bv.u[j] = vt[half * 8 + j];
        acc[ct] = wmma_f16(pf.v, bv.v, acc[ct]);
      }
    }
  }

  if (wave < 15) {
#pragma unroll
    for (int ct = 0; ct < 8; ++ct)
#pragma unroll
      for (int r = 0; r < 8; ++r) {
        int ml = r + 8 * half;
        int m2 = wave * 16 + ml;
        if (m2 < 225) {
          int ch = head * 128 + ct * 16 + l15;
          ao[((size_t)win * 225 + m2) * 512 + ch] = (_Float16)(acc[ct][r] / lrun[r]);
        }
      }
  }
}

// ---------------------------------------------------------------------------
// Kernel 3: projection GEMM.  ao (32400 x 512 f16) x W_proj (512x512 f32) + b
// -> f32 output.
// ---------------------------------------------------------------------------
__global__ __launch_bounds__(256) void proj_gemm_kernel(
    const _Float16* __restrict__ A, const float* __restrict__ W,
    const float* __restrict__ bias, float* __restrict__ out)
{
  __shared__ _Float16 s_w[64 * 32];
  const int tid = threadIdx.x, lane = tid & 31, wave = tid >> 5;
  const int half = lane >> 4, l15 = lane & 15;
  const int m0 = blockIdx.x * 128 + wave * 16;
  const int n0 = blockIdx.y * 64;
  const int Mtot = 32400;

  const int mrow = m0 + l15;
  const bool mval = mrow < Mtot;
  const u32* ar = (const u32*)(A + (size_t)(mval ? mrow : 0) * 512);

  Frag16 a[16];
#pragma unroll
  for (int kc = 0; kc < 16; ++kc)
#pragma unroll
    for (int j = 0; j < 8; ++j)
      a[kc].u[j] = mval ? ar[kc * 16 + (kbaseA(j, half) >> 1)] : 0u;

  v8f zero = {};
  v8f acc[4];
#pragma unroll
  for (int nt = 0; nt < 4; ++nt) acc[nt] = zero;

  for (int kc = 0; kc < 16; ++kc) {
    __syncthreads();
#pragma unroll
    for (int i = tid; i < 512; i += 256) {
      int kk = i >> 4;            // 0..31
      int nn = (i & 15) * 4;      // 0..60
      float4 w4 = *(const float4*)&W[(size_t)(kc * 32 + kk) * 512 + (n0 + nn)];
      s_w[(nn + 0) * 32 + kk] = (_Float16)w4.x;
      s_w[(nn + 1) * 32 + kk] = (_Float16)w4.y;
      s_w[(nn + 2) * 32 + kk] = (_Float16)w4.z;
      s_w[(nn + 3) * 32 + kk] = (_Float16)w4.w;
    }
    __syncthreads();
#pragma unroll
    for (int nt = 0; nt < 4; ++nt) {
      Frag16 b;
      const u32* wp = (const u32*)(s_w + (nt * 16 + l15) * 32);
#pragma unroll
      for (int j = 0; j < 8; ++j) b.u[j] = wp[half * 8 + j];
      acc[nt] = wmma_f16(a[kc].v, b.v, acc[nt]);
    }
  }

#pragma unroll
  for (int nt = 0; nt < 4; ++nt) {
    int n = n0 + nt * 16 + l15;
    float bn = bias[n];
#pragma unroll
    for (int r = 0; r < 8; ++r) {
      int m = m0 + r + 8 * half;
      if (m < Mtot) out[(size_t)m * 512 + n] = acc[nt][r] + bn;
    }
  }
}

// ---------------------------------------------------------------------------
extern "C" void kernel_launch(void* const* d_in, const int* in_sizes, int n_in,
                              void* d_out, int out_size, void* d_ws, size_t ws_size,
                              hipStream_t stream) {
  (void)in_sizes; (void)n_in; (void)out_size; (void)ws_size;
  const float* x0    = (const float*)d_in[0];
  const float* x1    = (const float*)d_in[1];
  const float* Wqkv  = (const float*)d_in[2];
  const float* bqkv  = (const float*)d_in[3];
  const float* Wproj = (const float*)d_in[4];
  const float* bproj = (const float*)d_in[5];
  float* out = (float*)d_out;

  char* ws = (char*)d_ws;
  const size_t szMG = (size_t)32400 * 512 * 2;  // f16 grid buffers
  const size_t szMP = (size_t)720 * 512 * 2;    // f16 pooled buffers
  _Float16* qg  = (_Float16*)ws; ws += szMG;
  _Float16* kgb = (_Float16*)ws; ws += szMG;
  _Float16* vgb = (_Float16*)ws; ws += szMG;
  _Float16* kp  = (_Float16*)ws; ws += szMP;
  _Float16* vp  = (_Float16*)ws; ws += szMP;
  _Float16* ao  = (_Float16*)ws; ws += szMG;

  // 1) grid QKV: 32400 rows, all 1536 cols
  qkv_gemm_kernel<<<dim3(254, 24), dim3(256), 0, stream>>>(
      x0, Wqkv, bqkv, qg, kgb, vgb, 32400, 0, 0);
  // 2) pooled QKV: 720 remapped rows, K/V cols only (n >= 512)
  qkv_gemm_kernel<<<dim3(6, 16), dim3(256), 0, stream>>>(
      x1, Wqkv, bqkv, nullptr, kp, vp, 720, 512, 1);
  // 3) attention per (window, head)
  attn_kernel<<<dim3(144, 4), dim3(512), 0, stream>>>(qg, kgb, vgb, kp, vp, ao);
  // 4) output projection
  proj_gemm_kernel<<<dim3(254, 8), dim3(256), 0, stream>>>(ao, Wproj, bproj, out);
}
